// GlobalLocalBlock_83700322664779
// MI455X (gfx1250) — matrix operand order repaired and checked by reading, not compiled, not run on hardware
//
#include <hip/hip_runtime.h>
#include <math.h>

// ---------------- problem constants ----------------
constexpr int B_   = 4, T_ = 8, H_ = 14, W_ = 14;
constexpr int DIM  = 768, NH = 8, HD = 96;
constexpr int G_   = 1;
constexpr int N_TOK = T_*H_*W_ + 1 + G_;   // 1570
constexpr int HID  = 4*DIM;                // 3072
constexpr int M_ROWS = B_*N_TOK;           // 6280
constexpr int MT32 = (M_ROWS + 31)/32;     // 197 M-supertiles (32 rows each)
constexpr int MP   = MT32*32;              // 6304 padded rows

typedef __attribute__((ext_vector_type(16))) __bf16         v16bf;
typedef __attribute__((ext_vector_type(8)))  float          v8f;
typedef __attribute__((ext_vector_type(8)))  unsigned short v8u;
typedef __attribute__((ext_vector_type(16))) unsigned short v16u;

#define DEV __device__ __forceinline__

DEV unsigned short f2bf(float f) {           // fp32 -> bf16, round-to-nearest-even
  unsigned u = __builtin_bit_cast(unsigned, f);
  unsigned r = (u + 0x7FFFu + ((u >> 16) & 1u)) >> 16;
  return (unsigned short)r;
}

DEV v16bf pack16(v8u lo, v8u hi) {
  v16u t;
#pragma unroll
  for (int i = 0; i < 8; ++i) { t[i] = lo[i]; t[i+8] = hi[i]; }
  return __builtin_bit_cast(v16bf, t);
}

// A fragment: two 16B chunks 16 elements apart (ISA 7.12.2 16-bit A 16x32)
DEV v16bf ldA(const unsigned short* p) {
  return pack16(*(const v8u*)p, *(const v8u*)(p + 16));
}
// B fragment: two 16B chunks 16 elements apart (same K order as A)
DEV v16bf ldB(const unsigned short* p) {
  return pack16(*(const v8u*)p, *(const v8u*)(p + 16));
}

// ---------------- fp32 -> bf16 weight conversion ----------------
__global__ void cvt_bf16_kernel(const float* __restrict__ src,
                                unsigned short* __restrict__ dst, int n) {
  int i = blockIdx.x*blockDim.x + threadIdx.x;
  if (i < n) dst[i] = f2bf(src[i]);
}

__global__ void zero_u16_kernel(unsigned short* __restrict__ p, int n) {
  int i = blockIdx.x*blockDim.x + threadIdx.x;
  if (i < n) p[i] = 0;
}

// ---------------- LayerNorm over DIM=768, bf16 output (padded rows zeroed) ---
__global__ void ln_bf16_kernel(const float* __restrict__ x,
                               const float* __restrict__ g,
                               const float* __restrict__ b,
                               unsigned short* __restrict__ out, int mrows) {
  const int row = blockIdx.x;          // grid = MP
  const int tid = threadIdx.x;         // 256 threads, 3 cols each
  if (row >= mrows) {
    out[(size_t)row*DIM + tid]       = 0;
    out[(size_t)row*DIM + tid + 256] = 0;
    out[(size_t)row*DIM + tid + 512] = 0;
    return;
  }
  const float* xr = x + (size_t)row*DIM;
  float v0 = xr[tid], v1 = xr[tid+256], v2 = xr[tid+512];
  __shared__ float s1[256], s2[256];
  s1[tid] = v0+v1+v2;
  s2[tid] = v0*v0 + v1*v1 + v2*v2;
  __syncthreads();
  for (int off = 128; off > 0; off >>= 1) {
    if (tid < off) { s1[tid] += s1[tid+off]; s2[tid] += s2[tid+off]; }
    __syncthreads();
  }
  const float mean = s1[0]*(1.f/DIM);
  const float var  = s2[0]*(1.f/DIM) - mean*mean;
  const float inv  = rsqrtf(var + 1e-5f);
  out[(size_t)row*DIM + tid]       = f2bf((v0-mean)*inv*g[tid]     + b[tid]);
  out[(size_t)row*DIM + tid + 256] = f2bf((v1-mean)*inv*g[tid+256] + b[tid+256]);
  out[(size_t)row*DIM + tid + 512] = f2bf((v2-mean)*inv*g[tid+512] + b[tid+512]);
}

// ---------------- bf16 WMMA GEMM: out[M,Nout] = A[M,K] @ W[Nout,K]^T + bias --
// Register-blocked 32(M) x 64(N) per wave (8 accumulators), software-pipelined:
// fragments for K-step k+1 are loaded while the 8 WMMAs of step k execute, so
// loads stay in flight behind the matrix pipe instead of serializing on waits.
enum { EPI_F32 = 0, EPI_RES = 1, EPI_GELU_BF16 = 2 };

template<int EPI>
__global__ void gemm_wmma_kernel(const unsigned short* __restrict__ A,   // [MP x K] bf16
                                 const unsigned short* __restrict__ Wt,  // [Nout x K] bf16
                                 const float* __restrict__ bias,         // [Nout]
                                 const float* __restrict__ resid,        // [M x Nout] (EPI_RES)
                                 float* __restrict__ outF,
                                 unsigned short* __restrict__ outBF,
                                 int M, int Nout, int K) {
  const int lane  = threadIdx.x & 31;
  const int wave  = threadIdx.x >> 5;
  const int l16   = lane & 15;
  const int half  = lane >> 4;
  const int nbase = blockIdx.x*256 + wave*64;   // first output column of this wave
  const int mbase = blockIdx.y*32;              // first output row of this wave

  const unsigned short* a0p = A  + (size_t)(mbase + l16)*K      + half*8;
  const unsigned short* a1p = a0p + (size_t)16*K;
  const unsigned short* b0p = Wt + (size_t)(nbase + l16)*K      + half*8;
  const unsigned short* b1p = b0p + (size_t)16*K;
  const unsigned short* b2p = b0p + (size_t)32*K;
  const unsigned short* b3p = b0p + (size_t)48*K;

  v8f acc[2][4];
#pragma unroll
  for (int i = 0; i < 2; ++i)
#pragma unroll
    for (int j = 0; j < 4; ++j) acc[i][j] = (v8f){};

  // prologue: fragments for K-step 0
  v16bf a0 = ldA(a0p), a1 = ldA(a1p);
  v16bf b0 = ldB(b0p), b1 = ldB(b1p), b2 = ldB(b2p), b3 = ldB(b3p);

  for (int kk = 32; kk < K; kk += 32) {
    // issue next step's loads first (independent of current WMMAs)
    const v16bf na0 = ldA(a0p + kk), na1 = ldA(a1p + kk);
    const v16bf nb0 = ldB(b0p + kk), nb1 = ldB(b1p + kk);
    const v16bf nb2 = ldB(b2p + kk), nb3 = ldB(b3p + kk);

    acc[0][0] = __builtin_amdgcn_wmma_f32_16x16x32_bf16(false, a0, false, b0, (short)0, acc[0][0], false, false);
    acc[1][0] = __builtin_amdgcn_wmma_f32_16x16x32_bf16(false, a1, false, b0, (short)0, acc[1][0], false, false);
    acc[0][1] = __builtin_amdgcn_wmma_f32_16x16x32_bf16(false, a0, false, b1, (short)0, acc[0][1], false, false);
    acc[1][1] = __builtin_amdgcn_wmma_f32_16x16x32_bf16(false, a1, false, b1, (short)0, acc[1][1], false, false);
    acc[0][2] = __builtin_amdgcn_wmma_f32_16x16x32_bf16(false, a0, false, b2, (short)0, acc[0][2], false, false);
    acc[1][2] = __builtin_amdgcn_wmma_f32_16x16x32_bf16(false, a1, false, b2, (short)0, acc[1][2], false, false);
    acc[0][3] = __builtin_amdgcn_wmma_f32_16x16x32_bf16(false, a0, false, b3, (short)0, acc[0][3], false, false);
    acc[1][3] = __builtin_amdgcn_wmma_f32_16x16x32_bf16(false, a1, false, b3, (short)0, acc[1][3], false, false);

    a0 = na0; a1 = na1; b0 = nb0; b1 = nb1; b2 = nb2; b3 = nb3;
  }
  // epilogue K-step
  acc[0][0] = __builtin_amdgcn_wmma_f32_16x16x32_bf16(false, a0, false, b0, (short)0, acc[0][0], false, false);
  acc[1][0] = __builtin_amdgcn_wmma_f32_16x16x32_bf16(false, a1, false, b0, (short)0, acc[1][0], false, false);
  acc[0][1] = __builtin_amdgcn_wmma_f32_16x16x32_bf16(false, a0, false, b1, (short)0, acc[0][1], false, false);
  acc[1][1] = __builtin_amdgcn_wmma_f32_16x16x32_bf16(false, a1, false, b1, (short)0, acc[1][1], false, false);
  acc[0][2] = __builtin_amdgcn_wmma_f32_16x16x32_bf16(false, a0, false, b2, (short)0, acc[0][2], false, false);
  acc[1][2] = __builtin_amdgcn_wmma_f32_16x16x32_bf16(false, a1, false, b2, (short)0, acc[1][2], false, false);
  acc[0][3] = __builtin_amdgcn_wmma_f32_16x16x32_bf16(false, a0, false, b3, (short)0, acc[0][3], false, false);
  acc[1][3] = __builtin_amdgcn_wmma_f32_16x16x32_bf16(false, a1, false, b3, (short)0, acc[1][3], false, false);

  // C/D layout: VGPR r -> row base + r (+8 for lanes 16..31), col = tile base + l16
#pragma unroll
  for (int j = 0; j < 4; ++j) {
    const int   col  = nbase + j*16 + l16;
    const float bcol = bias[col];
#pragma unroll
    for (int i = 0; i < 2; ++i) {
#pragma unroll
      for (int r = 0; r < 8; ++r) {
        const int m = mbase + i*16 + r + half*8;
        if (m >= M) continue;
        float v = acc[i][j][r] + bcol;
        const size_t idx = (size_t)m*Nout + col;
        if (EPI == EPI_RES) {
          outF[idx] = v + resid[idx];
        } else if (EPI == EPI_GELU_BF16) {
          outBF[idx] = f2bf(0.5f*v*(1.f + erff(v*0.70710678118f)));
        } else {
          outF[idx] = v;
        }
      }
    }
  }
}

// ---------------- depthwise 3x3x3 conv pool + LayerNorm over HD --------------
// grid = (N_TOK, B*NH, 3), block = 128 (96 active channels)
__global__ void pool_ln_kernel(const float* __restrict__ qkv,   // [B*N, 3*DIM]
                               const float* __restrict__ cwq, const float* __restrict__ gq, const float* __restrict__ bq,
                               const float* __restrict__ cwk, const float* __restrict__ gk, const float* __restrict__ bk,
                               const float* __restrict__ cwv, const float* __restrict__ gv, const float* __restrict__ bv,
                               float* __restrict__ pooled) {    // [3, B*NH, N, HD]
  const int n     = blockIdx.x;
  const int bh    = blockIdx.y;
  const int which = blockIdx.z;
  const int b     = bh >> 3;
  const int head  = bh & 7;
  const int tid   = threadIdx.x;

  const float* cw; const float* lg; const float* lb;
  if      (which == 0) { cw = cwq; lg = gq; lb = bq; }
  else if (which == 1) { cw = cwk; lg = gk; lb = bk; }
  else                 { cw = cwv; lg = gv; lb = bv; }

  float val = 0.f;
  if (tid < HD) {
    const size_t colBase = (size_t)which*DIM + head*HD + tid;
    if (n < G_ + 1) {                 // global + cls tokens pass through
      val = qkv[(size_t)(b*N_TOK + n)*(3*DIM) + colBase];
    } else {
      const int tok = n - (G_ + 1);
      const int t = tok / (H_*W_);
      const int rm = tok % (H_*W_);
      const int y = rm / W_;
      const int x = rm % W_;
      const float* wch = cw + tid*27; // conv shape (HD,1,3,3,3)
      float s = 0.f;
      for (int dt = -1; dt <= 1; ++dt) {
        const int tt = t + dt; if (tt < 0 || tt >= T_) continue;
        for (int dy = -1; dy <= 1; ++dy) {
          const int yy = y + dy; if (yy < 0 || yy >= H_) continue;
          for (int dx = -1; dx <= 1; ++dx) {
            const int xx = x + dx; if (xx < 0 || xx >= W_) continue;
            const int ntok = (tt*H_ + yy)*W_ + xx;
            s += wch[(dt+1)*9 + (dy+1)*3 + (dx+1)] *
                 qkv[(size_t)(b*N_TOK + G_ + 1 + ntok)*(3*DIM) + colBase];
          }
        }
      }
      val = s;
    }
  }
  __shared__ float s1[128], s2[128];
  s1[tid] = (tid < HD) ? val      : 0.f;
  s2[tid] = (tid < HD) ? val*val  : 0.f;
  __syncthreads();
  for (int off = 64; off > 0; off >>= 1) {
    if (tid < off) { s1[tid] += s1[tid+off]; s2[tid] += s2[tid+off]; }
    __syncthreads();
  }
  const float mean = s1[0]*(1.f/HD);
  const float var  = s2[0]*(1.f/HD) - mean*mean;
  const float inv  = rsqrtf(var + 1e-5f);
  if (tid < HD)
    pooled[(((size_t)which*(B_*NH) + bh)*N_TOK + n)*HD + tid] =
        (val - mean)*inv*lg[tid] + lb[tid];
}

// ---------------- masked attention (collapses to 2-term softmax) -------------
// mask: offset==0 only at j==i and j==0 (for i>=1); exp(-1e8) underflows to 0.
// One wave per (b, head, n); lane handles 3 of 96 channels.
__global__ void attn_kernel(const float* __restrict__ pooled,   // [3, B*NH, N, HD]
                            unsigned short* __restrict__ oBF) { // [B*N, DIM] bf16
  const int gwave = (int)((blockIdx.x*blockDim.x + threadIdx.x) >> 5);
  const int lane  = threadIdx.x & 31;
  if (gwave >= B_*NH*N_TOK) return;
  const int n  = gwave % N_TOK;
  const int bh = gwave / N_TOK;
  const int b    = bh >> 3;
  const int head = bh & 7;

  const float* q  = pooled + ((size_t)(0*(B_*NH) + bh)*N_TOK + n)*HD;
  const float* k  = pooled + ((size_t)(1*(B_*NH) + bh)*N_TOK)*HD;
  const float* v  = pooled + ((size_t)(2*(B_*NH) + bh)*N_TOK)*HD;
  const float* k0 = k;                 const float* kn = k + (size_t)n*HD;
  const float* v0 = v;                 const float* vn = v + (size_t)n*HD;

  const int c = lane*3;
  const float q0 = q[c], q1 = q[c+1], q2 = q[c+2];
  float s0 = q0*k0[c] + q1*k0[c+1] + q2*k0[c+2];
  float sn = q0*kn[c] + q1*kn[c+1] + q2*kn[c+2];
  for (int off = 16; off > 0; off >>= 1) {
    s0 += __shfl_xor(s0, off, 32);
    sn += __shfl_xor(sn, off, 32);
  }
  const size_t orow = (size_t)(b*N_TOK + n)*DIM + head*HD;
  if (n == 0) {                        // global token attends only itself
    oBF[orow + c]   = f2bf(v0[c]);
    oBF[orow + c+1] = f2bf(v0[c+1]);
    oBF[orow + c+2] = f2bf(v0[c+2]);
  } else {
    const float scale = rsqrtf((float)HD);
    s0 *= scale; sn *= scale;
    const float m  = fmaxf(s0, sn);
    const float e0 = expf(s0 - m), en = expf(sn - m);
    const float inv = 1.f/(e0 + en);
    const float w0 = e0*inv, wn = en*inv;
    oBF[orow + c]   = f2bf(w0*v0[c]   + wn*vn[c]);
    oBF[orow + c+1] = f2bf(w0*v0[c+1] + wn*vn[c+1]);
    oBF[orow + c+2] = f2bf(w0*v0[c+2] + wn*vn[c+2]);
  }
}

// ---------------- launcher ----------------
extern "C" void kernel_launch(void* const* d_in, const int* in_sizes, int n_in,
                              void* d_out, int out_size, void* d_ws, size_t ws_size,
                              hipStream_t stream) {
  (void)in_sizes; (void)n_in; (void)out_size; (void)ws_size;
  const float* x      = (const float*)d_in[0];
  const float* n1g    = (const float*)d_in[4];
  const float* n1b    = (const float*)d_in[5];
  const float* qkv_w  = (const float*)d_in[6];
  const float* qkv_b  = (const float*)d_in[7];
  const float* conv_q = (const float*)d_in[8];
  const float* nq_g   = (const float*)d_in[9];
  const float* nq_b   = (const float*)d_in[10];
  const float* conv_k = (const float*)d_in[11];
  const float* nk_g   = (const float*)d_in[12];
  const float* nk_b   = (const float*)d_in[13];
  const float* conv_v = (const float*)d_in[14];
  const float* nv_g   = (const float*)d_in[15];
  const float* nv_b   = (const float*)d_in[16];
  const float* proj_w = (const float*)d_in[17];
  const float* proj_b = (const float*)d_in[18];
  const float* n2g    = (const float*)d_in[19];
  const float* n2b    = (const float*)d_in[20];
  const float* fc1_w  = (const float*)d_in[21];
  const float* fc1_b  = (const float*)d_in[22];
  const float* fc2_w  = (const float*)d_in[23];
  const float* fc2_b  = (const float*)d_in[24];

  char* ws = (char*)d_ws;
  auto alloc = [&](size_t bytes) -> char* {
    char* p = ws; ws += (bytes + 255) & ~(size_t)255; return p;
  };
  unsigned short* xn   = (unsigned short*)alloc((size_t)MP*DIM*2);      // ln1 out (bf16)
  unsigned short* wq   = (unsigned short*)alloc((size_t)3*DIM*DIM*2);   // qkv_w bf16
  unsigned short* wp   = (unsigned short*)alloc((size_t)DIM*DIM*2);     // proj_w bf16
  unsigned short* w1   = (unsigned short*)alloc((size_t)HID*DIM*2);     // fc1_w bf16
  unsigned short* w2   = (unsigned short*)alloc((size_t)DIM*HID*2);     // fc2_w bf16
  float*          qkv  = (float*)alloc((size_t)M_ROWS*3*DIM*4);         // qkv activations
  float*          pool = (float*)alloc((size_t)3*B_*NH*N_TOK*HD*4);     // pooled q/k/v
  unsigned short* obf  = (unsigned short*)alloc((size_t)MP*DIM*2);      // attn out (bf16)
  float*          xres = (float*)alloc((size_t)M_ROWS*DIM*4);           // x + proj
  unsigned short* x2n  = (unsigned short*)alloc((size_t)MP*DIM*2);      // ln2 out (bf16)
  unsigned short* h1   = (unsigned short*)alloc((size_t)MP*HID*2);      // gelu(fc1) bf16

  const int CT = 256;
  auto nb = [](int n, int t) { return (n + t - 1)/t; };

  // weights -> bf16
  cvt_bf16_kernel<<<nb(3*DIM*DIM, CT), CT, 0, stream>>>(qkv_w, wq, 3*DIM*DIM);
  cvt_bf16_kernel<<<nb(DIM*DIM,   CT), CT, 0, stream>>>(proj_w, wp, DIM*DIM);
  cvt_bf16_kernel<<<nb(HID*DIM,   CT), CT, 0, stream>>>(fc1_w, w1, HID*DIM);
  cvt_bf16_kernel<<<nb(DIM*HID,   CT), CT, 0, stream>>>(fc2_w, w2, DIM*HID);

  // LN1 -> bf16 (zeros the pad rows)
  ln_bf16_kernel<<<MP, 256, 0, stream>>>(x, n1g, n1b, xn, M_ROWS);

  // QKV GEMM: [6280,768] x [2304,768]^T   grid: 2304/256 x 197
  gemm_wmma_kernel<EPI_F32><<<dim3((3*DIM)/256, MT32), 128, 0, stream>>>(
      xn, wq, qkv_b, nullptr, qkv, nullptr, M_ROWS, 3*DIM, DIM);

  // depthwise conv pool + LN per head
  pool_ln_kernel<<<dim3(N_TOK, B_*NH, 3), 128, 0, stream>>>(
      qkv, conv_q, nq_g, nq_b, conv_k, nk_g, nk_b, conv_v, nv_g, nv_b, pool);

  // masked attention (2-term softmax), output in [B,N,DIM] bf16
  zero_u16_kernel<<<nb((MP - M_ROWS)*DIM, CT), CT, 0, stream>>>(
      obf + (size_t)M_ROWS*DIM, (MP - M_ROWS)*DIM);
  attn_kernel<<<nb(B_*NH*N_TOK*32, CT), CT, 0, stream>>>(pool, obf);

  // proj GEMM + residual (x)
  gemm_wmma_kernel<EPI_RES><<<dim3(DIM/256, MT32), 128, 0, stream>>>(
      obf, wp, proj_b, x, xres, nullptr, M_ROWS, DIM, DIM);

  // LN2 -> bf16
  ln_bf16_kernel<<<MP, 256, 0, stream>>>(xres, n2g, n2b, x2n, M_ROWS);

  // FC1 GEMM + exact GELU, bf16 out
  gemm_wmma_kernel<EPI_GELU_BF16><<<dim3(HID/256, MT32), 128, 0, stream>>>(
      x2n, w1, fc1_b, nullptr, nullptr, h1, M_ROWS, HID, DIM);
  zero_u16_kernel<<<nb((MP - M_ROWS)*HID, CT), CT, 0, stream>>>(
      h1 + (size_t)M_ROWS*HID, (MP - M_ROWS)*HID);

  // FC2 GEMM + residual (xres) -> final output
  gemm_wmma_kernel<EPI_RES><<<dim3(DIM/256, MT32), 128, 0, stream>>>(
      h1, w2, fc2_b, xres, (float*)d_out, nullptr, M_ROWS, DIM, HID);
}
